// TriViewGATEncoder_16999480558010
// MI455X (gfx1250) — matrix operand order repaired and checked by reading, not compiled, not run on hardware
//
#include <hip/hip_runtime.h>
#include <hip/hip_bf16.h>
#include <math.h>

// ---------------------------------------------------------------------------
// Types for CDNA5 WMMA (wave32): D(16x16 f32) = A(16x32 bf16) x B(32x16 bf16) + C
// ---------------------------------------------------------------------------
typedef __attribute__((ext_vector_type(16))) __bf16 v16bf;
typedef __attribute__((ext_vector_type(8)))  float  v8f;

// Native convert: lets the backend use v_cvt_*bf16_f32 (packs 2 per VALU op).
__device__ __forceinline__ __bf16 f2bf(float f) { return (__bf16)f; }

// A fragment (16x32 bf16) from a row-major bf16 tile in LDS.
// ISA layout: lane m = lane&15, half = lane>>4; element i -> K = 16*(i>>3) + 8*half + (i&7),
// i.e. two contiguous 8-element runs -> two 16-byte LDS loads.
__device__ __forceinline__ v16bf load_a_lds(const __bf16* T, int ldt, int k0, int lane) {
    int m = lane & 15, half = lane >> 4;
    const __bf16* tp = T + m * ldt + k0 + 8 * half;
    v16bf a;
    __builtin_memcpy(&a, tp, 16);                    // K = k0+8*half   + [0..7]
    __builtin_memcpy((char*)&a + 16, tp + 16, 16);   // K = k0+16+8*half+ [0..7]
    return a;
}

// A fragment from row-major f32 global memory: four float4 loads + convert.
__device__ __forceinline__ v16bf load_a_g(const float* A, int lda, int row0, int k0, int lane) {
    int m = lane & 15, half = lane >> 4;
    const float* ap = A + (size_t)(row0 + m) * lda + k0 + 8 * half;
    float4 q0 = *(const float4*)(ap);
    float4 q1 = *(const float4*)(ap + 4);
    float4 q2 = *(const float4*)(ap + 16);
    float4 q3 = *(const float4*)(ap + 20);
    v16bf a;
    a[0] = f2bf(q0.x); a[1] = f2bf(q0.y); a[2] = f2bf(q0.z); a[3] = f2bf(q0.w);
    a[4] = f2bf(q1.x); a[5] = f2bf(q1.y); a[6] = f2bf(q1.z); a[7] = f2bf(q1.w);
    a[8] = f2bf(q2.x); a[9] = f2bf(q2.y); a[10] = f2bf(q2.z); a[11] = f2bf(q2.w);
    a[12] = f2bf(q3.x); a[13] = f2bf(q3.y); a[14] = f2bf(q3.z); a[15] = f2bf(q3.w);
    return a;
}

// B fragment: 32x16 tile of row-major f32 B[K,N], converted to bf16.
// ISA layout: lane n = lane&15, half = lane>>4; element i -> K = 16*(i>>3) + 8*half + (i&7).
// Loads are coalesced across lanes (n contiguous); ldb is constexpr at call sites.
template <int LDB>
__device__ __forceinline__ v16bf load_b_g(const float* B, int k0, int col0, int lane) {
    int n = lane & 15, half = lane >> 4;
    const float* bp = B + (size_t)(k0 + 8 * half) * LDB + col0 + n;
    v16bf b;
#pragma unroll
    for (int i = 0; i < 16; ++i) b[i] = f2bf(bp[(size_t)(i + (i & 8)) * LDB]);
    return b;
}

__device__ __forceinline__ v8f wmma_bf16(v16bf a, v16bf b, v8f c) {
    return __builtin_amdgcn_wmma_f32_16x16x32_bf16(false, a, false, b, (short)0, c, false, false);
}

// ---------------------------------------------------------------------------
// GEMM: C[M,Nc] = A[M,K] @ B[K,Nc] (f32 in/out, bf16 WMMA, f32 accum).
// Block = TN waves handles a 16-row stripe; A tile staged once in LDS as bf16
// and shared by all TN column tiles. Grid = M/16 (M multiple of 16).
// ---------------------------------------------------------------------------
template <int K, int TN>
__global__ __launch_bounds__(32 * TN) void gemm_bf16_wmma(const float* __restrict__ A,
                                                          const float* __restrict__ B,
                                                          float* __restrict__ C) {
    constexpr int Nc = TN * 16;
    constexpr int NT = 32 * TN;
    __shared__ __bf16 As[16 * K];

    int row0 = blockIdx.x << 4;
    int tid  = threadIdx.x;

    // Cooperative load: 16 rows x K cols is contiguous in row-major A.
    const float* abase = A + (size_t)row0 * K;
#pragma unroll
    for (int i = tid * 4; i < 16 * K; i += NT * 4) {
        float4 q = *(const float4*)(abase + i);
        union { __bf16 b[4]; uint2 u; } pk;
        pk.b[0] = f2bf(q.x); pk.b[1] = f2bf(q.y); pk.b[2] = f2bf(q.z); pk.b[3] = f2bf(q.w);
        *(uint2*)(As + i) = pk.u;
    }
    __syncthreads();

    int lane = tid & 31;
    int col0 = (tid >> 5) << 4;    // wave -> 16-column tile

    v8f c = {};
#pragma unroll
    for (int k0 = 0; k0 < K; k0 += 32) {
        v16bf a = load_a_lds(As, K, k0, lane);
        v16bf b = load_b_g<Nc>(B, k0, col0, lane);
        c = wmma_bf16(a, b, c);
    }

    int half = lane >> 4, nl = lane & 15;
    float* cp = C + (size_t)(row0 + 8 * half) * Nc + col0 + nl;
#pragma unroll
    for (int r = 0; r < 8; ++r) cp[(size_t)r * Nc] = c[r];
}

// ---------------------------------------------------------------------------
// Per-node attention logits: as[n,h] = sum_c H[n,h*CH+c]*a_src[h*CH+c] (and a_dst).
// ---------------------------------------------------------------------------
template <int HEADS, int CH>
__global__ void alpha_kernel(const float* __restrict__ H,
                             const float* __restrict__ a_src,
                             const float* __restrict__ a_dst,
                             float* __restrict__ as, float* __restrict__ ad, int N) {
    int idx = blockIdx.x * blockDim.x + threadIdx.x;
    if (idx >= N * HEADS) return;
    int n = idx / HEADS, h = idx % HEADS;
    const float* hp = H + (size_t)n * HEADS * CH + h * CH;
    float s = 0.f, d = 0.f;
#pragma unroll
    for (int c = 0; c < CH; ++c) {
        float v = hp[c];
        s += v * a_src[h * CH + c];
        d += v * a_dst[h * CH + c];
    }
    as[idx] = s;
    ad[idx] = d;
}

__global__ void fill_kernel(float* __restrict__ p, float v, size_t n) {
    size_t i = (size_t)blockIdx.x * blockDim.x + threadIdx.x;
    if (i < n) p[i] = v;
}

__device__ __forceinline__ void atomicMaxFloat(float* addr, float val) {
    if (val >= 0.f) atomicMax((int*)addr, __float_as_int(val));
    else            atomicMin((unsigned int*)addr, (unsigned int)__float_as_int(val));
}

__device__ __forceinline__ float lrelu(float v) { return v > 0.f ? v : 0.2f * v; }

// Pass 1: segment max over dst of leaky_relu(as[src]+ad[dst]).
template <int HEADS>
__global__ void edge_max_kernel(const int* __restrict__ src, const int* __restrict__ dst,
                                const float* __restrict__ as, const float* __restrict__ ad,
                                float* __restrict__ m, int E, int N) {
    int e = blockIdx.x * blockDim.x + threadIdx.x;
    if (e >= E + N) return;
    int s = (e < E) ? src[e] : (e - E);
    int d = (e < E) ? dst[e] : (e - E);
#pragma unroll
    for (int h = 0; h < HEADS; ++h)
        atomicMaxFloat(&m[d * HEADS + h], lrelu(as[s * HEADS + h] + ad[d * HEADS + h]));
}

// Pass 2: segment sum of exp(e - m[dst]).
template <int HEADS>
__global__ void edge_sum_kernel(const int* __restrict__ src, const int* __restrict__ dst,
                                const float* __restrict__ as, const float* __restrict__ ad,
                                const float* __restrict__ m, float* __restrict__ den,
                                int E, int N) {
    int e = blockIdx.x * blockDim.x + threadIdx.x;
    if (e >= E + N) return;
    int s = (e < E) ? src[e] : (e - E);
    int d = (e < E) ? dst[e] : (e - E);
#pragma unroll
    for (int h = 0; h < HEADS; ++h) {
        float v = lrelu(as[s * HEADS + h] + ad[d * HEADS + h]);
        atomicAdd(&den[d * HEADS + h], __expf(v - m[d * HEADS + h]));
    }
}

// Pass 3: acc[dst,f] += H[src,f] * alpha(e,head(f)).  One wave32 per edge;
// lanes stride the feature dim so gathers/atomics are coalesced 128B packets.
template <int HEADS, int F>
__global__ __launch_bounds__(256) void edge_scatter_kernel(
        const int* __restrict__ src, const int* __restrict__ dst,
        const float* __restrict__ as, const float* __restrict__ ad,
        const float* __restrict__ m, const float* __restrict__ den,
        const float* __restrict__ Hf, float* __restrict__ acc, int E, int N) {
    int lane = threadIdx.x & 31;
    int e = blockIdx.x * (blockDim.x >> 5) + (threadIdx.x >> 5);
    if (e >= E + N) return;
    int s = (e < E) ? src[e] : (e - E);
    int d = (e < E) ? dst[e] : (e - E);
    const float* hs = Hf + (size_t)s * F;
    float* ao = acc + (size_t)d * F;
#pragma unroll
    for (int j = 0; j < F / 32; ++j) {
        int f = j * 32 + lane;
        int h = f / (F / HEADS);                 // wave-uniform per j
        float v = lrelu(as[s * HEADS + h] + ad[d * HEADS + h]);
        float alpha = __expf(v - m[d * HEADS + h]) / (den[d * HEADS + h] + 1e-16f);
        atomicAdd(&ao[f], hs[f] * alpha);
    }
}

// out[n,f] = elu(acc[n,f] + b[f])
template <int F>
__global__ void finish_elu_kernel(const float* __restrict__ acc, const float* __restrict__ b,
                                  float* __restrict__ out, int N) {
    size_t i = (size_t)blockIdx.x * blockDim.x + threadIdx.x;
    if (i >= (size_t)N * F) return;
    float x = acc[i] + b[i % F];
    out[i] = x > 0.f ? x : expm1f(x);
}

// ---------------------------------------------------------------------------
// Fused projection: out = relu(h@pw1+pb1) @ pw2 + pb2   (h: [N,64])
// 128 threads = 4 waves per block, 16 rows per block; intermediate tile is
// round-tripped through LDS as bf16 so stage-2 can load it in A-fragment order.
// ---------------------------------------------------------------------------
__global__ __launch_bounds__(128) void proj_kernel(const float* __restrict__ h,
                                                   const float* __restrict__ pw1,
                                                   const float* __restrict__ pb1,
                                                   const float* __restrict__ pw2,
                                                   const float* __restrict__ pb2,
                                                   float* __restrict__ out, int M) {
    __shared__ __bf16 t[16 * 64];
    int lane = threadIdx.x & 31;
    int wave = threadIdx.x >> 5;                 // 0..3 -> 16-column tile
    int row0 = blockIdx.x << 4;
    if (row0 >= M) return;
    int half = lane >> 4, nl = lane & 15;

    // stage 1: t = relu(h_tile @ pw1 + pb1)
    v8f c = {};
#pragma unroll
    for (int k0 = 0; k0 < 64; k0 += 32) {
        v16bf a = load_a_g(h, 64, row0, k0, lane);
        v16bf b = load_b_g<64>(pw1, k0, wave * 16, lane);
        c = wmma_bf16(a, b, c);
    }
#pragma unroll
    for (int r = 0; r < 8; ++r) {
        int mm = r + 8 * half, nn = wave * 16 + nl;
        float v = c[r] + pb1[nn];
        t[mm * 64 + nn] = f2bf(v > 0.f ? v : 0.f);
    }
    __syncthreads();

    // stage 2: out = t @ pw2 + pb2
    v8f c2 = {};
#pragma unroll
    for (int k0 = 0; k0 < 64; k0 += 32) {
        v16bf a = load_a_lds(t, 64, k0, lane);
        v16bf b = load_b_g<64>(pw2, k0, wave * 16, lane);
        c2 = wmma_bf16(a, b, c2);
    }
#pragma unroll
    for (int r = 0; r < 8; ++r) {
        int mm = r + 8 * half, nn = wave * 16 + nl;
        out[(size_t)(row0 + mm) * 64 + nn] = c2[r] + pb2[nn];
    }
}

// ---------------------------------------------------------------------------
// Host launcher
// ---------------------------------------------------------------------------
extern "C" void kernel_launch(void* const* d_in, const int* in_sizes, int n_in,
                              void* d_out, int out_size, void* d_ws, size_t ws_size,
                              hipStream_t stream) {
    const float* x      = (const float*)d_in[0];
    const int*   eidx   = (const int*)d_in[1];
    const float* W1     = (const float*)d_in[2];
    const float* a_src1 = (const float*)d_in[3];
    const float* a_dst1 = (const float*)d_in[4];
    const float* b1     = (const float*)d_in[5];
    const float* W2     = (const float*)d_in[6];
    const float* a_src2 = (const float*)d_in[7];
    const float* a_dst2 = (const float*)d_in[8];
    const float* b2     = (const float*)d_in[9];
    const float* pw1    = (const float*)d_in[10];
    const float* pb1    = (const float*)d_in[11];
    const float* pw2    = (const float*)d_in[12];
    const float* pb2    = (const float*)d_in[13];
    float* out = (float*)d_out;

    const int N  = in_sizes[0] / 128;
    const int E  = in_sizes[1] / 2;
    const int ET = E + N;                        // with self loops
    const int* src = eidx;
    const int* dst = eidx + E;

    // workspace bump allocator (~81 MB total for N=50k)
    float* ws = (float*)d_ws;
    float* H1   = ws; ws += (size_t)N * 128;     // layer-1 features, reused for elu(out1)
    float* acc1 = ws; ws += (size_t)N * 128;
    float* as1  = ws; ws += (size_t)N * 4;
    float* ad1  = ws; ws += (size_t)N * 4;
    float* m1   = ws; ws += (size_t)N * 4;
    float* den1 = ws; ws += (size_t)N * 4;
    float* H2   = ws; ws += (size_t)N * 64;      // layer-2 features, reused for elu(out2)
    float* acc2 = ws; ws += (size_t)N * 64;
    float* as2  = ws; ws += (size_t)N;
    float* ad2  = ws; ws += (size_t)N;
    float* m2   = ws; ws += (size_t)N;
    float* den2 = ws; ws += (size_t)N;

    const int TB = 256;
    auto blocks  = [](size_t n, int tb) { return (unsigned)((n + tb - 1) / tb); };

    // ------------------ Layer 1 (4 heads x 32 ch) ------------------
    gemm_bf16_wmma<128, 8><<<N / 16, 256, 0, stream>>>(x, W1, H1);
    alpha_kernel<4, 32><<<blocks((size_t)N * 4, TB), TB, 0, stream>>>(H1, a_src1, a_dst1, as1, ad1, N);

    fill_kernel<<<blocks((size_t)N * 4, TB), TB, 0, stream>>>(m1, -3.0e38f, (size_t)N * 4);
    fill_kernel<<<blocks((size_t)N * 4, TB), TB, 0, stream>>>(den1, 0.f, (size_t)N * 4);
    fill_kernel<<<blocks((size_t)N * 128, TB), TB, 0, stream>>>(acc1, 0.f, (size_t)N * 128);

    edge_max_kernel<4><<<blocks(ET, TB), TB, 0, stream>>>(src, dst, as1, ad1, m1, E, N);
    edge_sum_kernel<4><<<blocks(ET, TB), TB, 0, stream>>>(src, dst, as1, ad1, m1, den1, E, N);
    edge_scatter_kernel<4, 128><<<blocks((size_t)ET * 32, TB), TB, 0, stream>>>(
        src, dst, as1, ad1, m1, den1, H1, acc1, E, N);
    finish_elu_kernel<128><<<blocks((size_t)N * 128, TB), TB, 0, stream>>>(acc1, b1, H1, N);

    // ------------------ Layer 2 (1 head x 64 ch) ------------------
    gemm_bf16_wmma<128, 4><<<N / 16, 128, 0, stream>>>(H1, W2, H2);
    alpha_kernel<1, 64><<<blocks(N, TB), TB, 0, stream>>>(H2, a_src2, a_dst2, as2, ad2, N);

    fill_kernel<<<blocks(N, TB), TB, 0, stream>>>(m2, -3.0e38f, (size_t)N);
    fill_kernel<<<blocks(N, TB), TB, 0, stream>>>(den2, 0.f, (size_t)N);
    fill_kernel<<<blocks((size_t)N * 64, TB), TB, 0, stream>>>(acc2, 0.f, (size_t)N * 64);

    edge_max_kernel<1><<<blocks(ET, TB), TB, 0, stream>>>(src, dst, as2, ad2, m2, E, N);
    edge_sum_kernel<1><<<blocks(ET, TB), TB, 0, stream>>>(src, dst, as2, ad2, m2, den2, E, N);
    edge_scatter_kernel<1, 64><<<blocks((size_t)ET * 32, TB), TB, 0, stream>>>(
        src, dst, as2, ad2, m2, den2, H2, acc2, E, N);
    finish_elu_kernel<64><<<blocks((size_t)N * 64, TB), TB, 0, stream>>>(acc2, b2, H2, N);

    // ------------------ Fused projection MLP ------------------
    proj_kernel<<<N / 16, 128, 0, stream>>>(H2, pw1, pb1, pw2, pb2, out, N);
}
